// PyGCardinalityHead_764504178969
// MI455X (gfx1250) — matrix operand order repaired and checked by reading, not compiled, not run on hardware
//
#include <hip/hip_runtime.h>
#include <hip/hip_bf16.h>
#include <math.h>

// ---------------------------------------------------------------------------
// Problem constants (from reference)
// ---------------------------------------------------------------------------
#define BB   512      // batch
#define CIN  200      // input channels
#define MM   128      // nodes per tree
#define N3   381      // 3*(M-1)
#define H1   512
#define H2   512
#define FF   256
#define GH   128      // F/2 gate hidden

typedef __attribute__((ext_vector_type(16))) __bf16 v16bf;
typedef __attribute__((ext_vector_type(8)))  float  v8f;

union Frag16 { v16bf v; uint4 q[2]; };

__device__ __forceinline__ __bf16 f2bf(float f) { return (__bf16)f; }

// ---------------------------------------------------------------------------
// Pack an (O x C) slice A_k[o][c] = w[o][c][k]  (w is (O,C,3) row-major, f32)
// into the WMMA bf16 A-fragment layout:
//   pk[(((k*(O/16)+tr)*steps + st)*32 + lane)*16 + s]
//   lane<16 : M=lane,    c_local = s<8 ? s   : s+8
//   lane>=16: M=lane-16, c_local = s<8 ? s+8 : s+16
// ---------------------------------------------------------------------------
__global__ void pack_wA(const float* __restrict__ w, __bf16* __restrict__ pk,
                        int O, int C, int steps) {
  int e = blockIdx.x * 256 + threadIdx.x;
  int total = 3 * (O >> 4) * steps * 512;
  if (e >= total) return;
  int s    = e & 15;
  int lane = (e >> 4) & 31;
  int st   = (e >> 9) % steps;
  int tmp  = (e >> 9) / steps;
  int tr   = tmp % (O >> 4);
  int k    = tmp / (O >> 4);
  int m, cl;
  if (lane < 16) { m = lane;      cl = (s < 8) ? s      : s + 8;  }
  else           { m = lane - 16; cl = (s < 8) ? s + 8  : s + 16; }
  int c = st * 32 + cl;
  float v = 0.f;
  if (c < C) v = w[((size_t)(tr * 16 + m) * C + c) * 3 + k];
  pk[e] = f2bf(v);
}

// ---------------------------------------------------------------------------
// Pack w (N x K, row-major) as B-fragments for B = w^T (K x N)
// ---------------------------------------------------------------------------
__global__ void pack_wB(const float* __restrict__ w, __bf16* __restrict__ pk,
                        int N, int K) {
  int steps = K >> 5;
  int e = blockIdx.x * 256 + threadIdx.x;
  int total = (N >> 4) * steps * 512;
  if (e >= total) return;
  int s    = e & 15;
  int lane = (e >> 4) & 31;
  int st   = (e >> 9) % steps;
  int nt   = (e >> 9) / steps;
  int nn, kk;
  if (lane < 16) { nn = lane;      kk = st * 32 + s;      }
  else           { nn = lane - 16; kk = st * 32 + 16 + s; }
  pk[e] = f2bf(w[(size_t)(nt * 16 + nn) * K + kk]);
}

// ---------------------------------------------------------------------------
// Fused tree-conv layer (STEPS = K-loop trip count, compile-time so the loop
// fully unrolls and all loads use immediate offsets):
//   optional LN(prev stats)+ReLU applied while staging src[b] -> LDS bf16
//   P_k = A_k x data_b  via v_wmma_f32_16x16x32_bf16
//   gather epilogue: y[b,o,n] = sum_k P_k[o, idx[3(n-1)+k]] + bias[o], col0=0
//   per-batch (sum, sumsq) accumulated atomically for the next layer's LN
// Grid: B * (O/64) workgroups, 256 threads (8 wave32).
// Dynamic LDS: max(128*CP*2 bytes data, 3*64*132*4 bytes P) -- aliased.
// ---------------------------------------------------------------------------
template <int STEPS>
__global__ __launch_bounds__(256)
void tree_conv_kernel(const float* __restrict__ src,      // (B, C, 128)
                      const float* __restrict__ stats_in, // (B,2) or null
                      const int*   __restrict__ idx,      // (B, 381)
                      const __bf16* __restrict__ pk,      // packed A frags
                      const float* __restrict__ bias,     // (O)
                      float* __restrict__ dst,            // (B, O, 128)
                      float* __restrict__ stats_out,      // (B,2) or null
                      int C, int O, int prevO) {
  constexpr int KC = STEPS * 32;
  constexpr int CP = KC + 8;

  extern __shared__ char smem[];
  __bf16* dsh = (__bf16*)smem;   // [128][CP]   node-major bf16 input
  float*  psh = (float*)smem;    // [3][64][132] P tiles (aliased after barrier)
  __shared__ int idx_s[384];

  const int tid  = threadIdx.x;
  const int lane = tid & 31;
  const int wave = tid >> 5;
  const int nstripes = O >> 6;
  const int b  = blockIdx.x / nstripes;
  const int rs = blockIdx.x % nstripes;

  // LayerNorm params of the *input* (previous layer's stats)
  float mean = 0.f, scale = 1.f;
  const bool ln = (stats_in != nullptr);
  if (ln) {
    float n  = (float)prevO * 128.f;
    float s  = stats_in[2 * b];
    float ss = stats_in[2 * b + 1];
    mean = s / n;
    float var = (ss - s * s / n) / (n - 1.f);
    scale = 1.f / (sqrtf(fmaxf(var, 0.f)) + 1e-5f);
  }

  // stage gather indices
  for (int i = tid; i < N3; i += 256) idx_s[i] = idx[(size_t)b * N3 + i];

  // stage input transposed (float4 along nodes): dsh[n][c] = lnrelu(src[b][c][n])
  const float* srcb = src + (size_t)b * C * MM;
  for (int e4 = tid; e4 < (MM / 4) * KC; e4 += 256) {
    int c  = e4 >> 5;            // MM/4 == 32
    int n4 = e4 & 31;
    float4 v = make_float4(0.f, 0.f, 0.f, 0.f);
    if (c < C) {
      v = *(const float4*)(srcb + (size_t)c * MM + n4 * 4);
      if (ln) {
        v.x = fmaxf((v.x - mean) * scale, 0.f);
        v.y = fmaxf((v.y - mean) * scale, 0.f);
        v.z = fmaxf((v.z - mean) * scale, 0.f);
        v.w = fmaxf((v.w - mean) * scale, 0.f);
      }
    }
    const int n = n4 * 4;
    dsh[(size_t)(n + 0) * CP + c] = f2bf(v.x);
    dsh[(size_t)(n + 1) * CP + c] = f2bf(v.y);
    dsh[(size_t)(n + 2) * CP + c] = f2bf(v.z);
    dsh[(size_t)(n + 3) * CP + c] = f2bf(v.w);
  }
  __syncthreads();

  // ---- WMMA main loop: wave -> (row tile rt, column half ch) ----
  const int rt  = wave & 3;            // 4 row tiles of the 64-row stripe
  const int ch  = wave >> 2;           // 0 or 1: 64-column half
  const int trg = rs * 4 + rt;         // global row tile
  v8f acc[3][4];
#pragma unroll
  for (int k = 0; k < 3; ++k)
#pragma unroll
    for (int ct = 0; ct < 4; ++ct) acc[k][ct] = (v8f)0.f;

  // hoisted per-lane base pointers; loop offsets become immediates
  const __bf16* apk[3];
#pragma unroll
  for (int k = 0; k < 3; ++k)
    apk[k] = pk + ((size_t)(k * (O >> 4) + trg) * STEPS << 9) + (lane << 4);
  const int b_n  = ch * 64 + (lane & 15);
  const int b_c0 = (lane & 16) ? 8 : 0;
  const __bf16* bpc[4];
#pragma unroll
  for (int ct = 0; ct < 4; ++ct)
    bpc[ct] = dsh + (size_t)(b_n + ct * 16) * CP + b_c0;

#pragma unroll
  for (int st = 0; st < STEPS; ++st) {
    Frag16 a[3];
#pragma unroll
    for (int k = 0; k < 3; ++k) {
      const __bf16* ap = apk[k] + st * 512;
      a[k].q[0] = *(const uint4*)(ap);
      a[k].q[1] = *(const uint4*)(ap + 8);
    }
    Frag16 bf[4];
#pragma unroll
    for (int ct = 0; ct < 4; ++ct) {
      const __bf16* bp = bpc[ct] + st * 32;
      bf[ct].q[0] = *(const uint4*)(bp);
      bf[ct].q[1] = *(const uint4*)(bp + 16);
    }
#pragma unroll
    for (int k = 0; k < 3; ++k)
#pragma unroll
      for (int ct = 0; ct < 4; ++ct)
        acc[k][ct] = __builtin_amdgcn_wmma_f32_16x16x32_bf16(
            false, a[k].v, false, bf[ct].v, (short)0, acc[k][ct], false, false);
  }
  __syncthreads();   // all waves done reading dsh -> safe to alias with psh

  // spill P tiles to LDS (C/D layout: vgpr v, lane<16 -> M=v, else M=v+8)
  const int prow = rt * 16 + ((lane & 16) ? 8 : 0);
  const int pcol = ch * 64 + (lane & 15);
#pragma unroll
  for (int k = 0; k < 3; ++k)
#pragma unroll
    for (int ct = 0; ct < 4; ++ct)
#pragma unroll
      for (int v = 0; v < 8; ++v)
        psh[(size_t)(k * 64 + prow + v) * 132 + pcol + ct * 16] = acc[k][ct][v];
  __syncthreads();

  // ---- gather epilogue (float4 stores) + LN statistics ----
  float lsum = 0.f, lss = 0.f;
  float* dstb = dst + ((size_t)b * O + rs * 64) * MM;
#pragma unroll
  for (int i = 0; i < 8; ++i) {
    int e4 = tid + i * 256;            // 2048 quads = 64 rows x 32 quads
    int ol = e4 >> 5;
    int n4 = e4 & 31;
    float vb = bias[rs * 64 + ol];
    float vv[4];
#pragma unroll
    for (int jj = 0; jj < 4; ++jj) {
      int n = n4 * 4 + jj;
      float t = 0.f;
      if (n != 0) {
        int p = 3 * (n - 1);
        t = psh[(size_t)(0 * 64 + ol) * 132 + idx_s[p + 0]]
          + psh[(size_t)(1 * 64 + ol) * 132 + idx_s[p + 1]]
          + psh[(size_t)(2 * 64 + ol) * 132 + idx_s[p + 2]]
          + vb;
      }
      vv[jj] = t;
      lsum += t;
      lss  += t * t;
    }
    *(float4*)(dstb + (size_t)ol * MM + n4 * 4) =
        make_float4(vv[0], vv[1], vv[2], vv[3]);
  }
  if (stats_out) {
#pragma unroll
    for (int off = 16; off; off >>= 1) {
      lsum += __shfl_down(lsum, off);
      lss  += __shfl_down(lss,  off);
    }
    if (lane == 0) {
      atomicAdd(&stats_out[2 * b],     lsum);
      atomicAdd(&stats_out[2 * b + 1], lss);
    }
  }
}

// ---------------------------------------------------------------------------
// Attention pooling: one workgroup per batch.
//   et = emb^T (128 x 256), gate1 = relu(et x gw1^T) via WMMA,
//   gate2 = gate1 x gw2^T, softmax over nodes, pool = attn^T x et,
//   combined[b] = [emb[b,:,1], pool]  written into d_out+512.
// Dynamic LDS: et bf16 [128][264] then gate f32 [128][132].
// ---------------------------------------------------------------------------
__global__ __launch_bounds__(256)
void attn_pool_kernel(const float* __restrict__ emb,   // (B, 256, 128)
                      const __bf16* __restrict__ pgw1, // packed B frags
                      const float* __restrict__ gb1,   // (128)
                      const float* __restrict__ gw2,   // (1,128)
                      const float* __restrict__ gb2,   // (1)
                      float* __restrict__ comb) {      // (B, 512) at d_out+512
  extern __shared__ char smem[];
  __bf16* et   = (__bf16*)smem;                       // [128][264]
  float*  gate = (float*)(smem + 128 * 264 * 2);      // [128][132]
  __shared__ float g2[128];
  __shared__ float sm_max, sm_sum;

  const int tid  = threadIdx.x;
  const int lane = tid & 31;
  const int wave = tid >> 5;
  const int b    = blockIdx.x;
  const int CP2  = 264;

  const float* embb = emb + (size_t)b * FF * MM;
  for (int e4 = tid; e4 < FF * (MM / 4); e4 += 256) {
    int f  = e4 >> 5;
    int n4 = e4 & 31;
    float4 v = *(const float4*)(embb + (size_t)f * MM + n4 * 4);
    const int n = n4 * 4;
    et[(size_t)(n + 0) * CP2 + f] = f2bf(v.x);
    et[(size_t)(n + 1) * CP2 + f] = f2bf(v.y);
    et[(size_t)(n + 2) * CP2 + f] = f2bf(v.z);
    et[(size_t)(n + 3) * CP2 + f] = f2bf(v.w);
  }
  __syncthreads();

  // GEMM: nodes(128) x F(256) x heads(128); wave = row tile
  const int rt = wave;
  v8f acc[8];
#pragma unroll
  for (int ct = 0; ct < 8; ++ct) acc[ct] = (v8f)0.f;

  const __bf16* ap0 =
      et + (size_t)(rt * 16 + (lane & 15)) * CP2 + ((lane & 16) ? 16 : 0);
  const __bf16* bp0 = pgw1 + (lane << 4);

#pragma unroll
  for (int st = 0; st < 8; ++st) {         // K = 256
    Frag16 a;
    const __bf16* ap = ap0 + st * 32;
    a.q[0] = *(const uint4*)(ap);
    a.q[1] = *(const uint4*)(ap + 8);
    Frag16 bf[8];
#pragma unroll
    for (int ct = 0; ct < 8; ++ct) {
      const __bf16* bp = bp0 + ((ct * 8 + st) << 9);
      bf[ct].q[0] = *(const uint4*)(bp);
      bf[ct].q[1] = *(const uint4*)(bp + 8);
    }
#pragma unroll
    for (int ct = 0; ct < 8; ++ct)
      acc[ct] = __builtin_amdgcn_wmma_f32_16x16x32_bf16(
          false, a.v, false, bf[ct].v, (short)0, acc[ct], false, false);
  }

  // relu + bias, store gate to LDS
  const int prow = rt * 16 + ((lane & 16) ? 8 : 0);
  const int pcol = lane & 15;
#pragma unroll
  for (int ct = 0; ct < 8; ++ct) {
    float bh = gb1[ct * 16 + pcol];
#pragma unroll
    for (int v = 0; v < 8; ++v)
      gate[(size_t)(prow + v) * 132 + ct * 16 + pcol] =
          fmaxf(acc[ct][v] + bh, 0.f);
  }
  __syncthreads();

  // gate2[n] = gate[n,:] . gw2 + gb2
  if (tid < MM) {
    float s = gb2[0];
    for (int h = 0; h < GH; ++h) s += gate[(size_t)tid * 132 + h] * gw2[h];
    g2[tid] = s;
  }
  __syncthreads();

  // softmax over 128 nodes
  if (tid < 32) {
    float mx = -1e30f;
    for (int i = lane; i < MM; i += 32) mx = fmaxf(mx, g2[i]);
#pragma unroll
    for (int o = 16; o; o >>= 1) mx = fmaxf(mx, __shfl_down(mx, o));
    mx = __shfl(mx, 0);
    float se = 0.f;
    for (int i = lane; i < MM; i += 32) se += expf(g2[i] - mx);
#pragma unroll
    for (int o = 16; o; o >>= 1) se += __shfl_down(se, o);
    if (lane == 0) { sm_max = mx; sm_sum = se; }
  }
  __syncthreads();
  if (tid < MM) g2[tid] = expf(g2[tid] - sm_max) / sm_sum;
  __syncthreads();

  // pool[f] = sum_n attn[n]*et[n][f];  combined = [root(f32), pool]
  const int f = tid;                      // 256 threads == F
  float p = 0.f;
  for (int n = 0; n < MM; ++n)
    p += g2[n] * (float)et[(size_t)n * CP2 + f];
  comb[(size_t)b * (2 * FF) + FF + f] = p;
  comb[(size_t)b * (2 * FF) + f]      = embb[(size_t)f * MM + 1];  // root f32
}

// ---------------------------------------------------------------------------
// Simple dense layers for the tiny readout MLP (VALU f32; ~76 MFLOP total)
// ---------------------------------------------------------------------------
__global__ void fc_kernel(const float* __restrict__ x, const float* __restrict__ w,
                          const float* __restrict__ bias, float* __restrict__ y,
                          int IN, int OUT, int rows, int relu) {
  int e = blockIdx.x * blockDim.x + threadIdx.x;
  if (e >= rows * OUT) return;
  int r = e / OUT, o = e % OUT;
  const float* xr = x + (size_t)r * IN;
  const float* wr = w + (size_t)o * IN;
  float a = bias[o];
  for (int i = 0; i < IN; ++i) a = fmaf(xr[i], wr[i], a);
  y[e] = relu ? fmaxf(a, 0.f) : a;
}

// ---------------------------------------------------------------------------
// Host launcher
// ---------------------------------------------------------------------------
extern "C" void kernel_launch(void* const* d_in, const int* in_sizes, int n_in,
                              void* d_out, int out_size, void* d_ws, size_t ws_size,
                              hipStream_t stream) {
  (void)in_sizes; (void)n_in; (void)out_size; (void)ws_size;

  const float* trees = (const float*)d_in[0];
  const int*   idx   = (const int*)  d_in[1];
  const float* w1    = (const float*)d_in[2];
  const float* b1    = (const float*)d_in[3];
  const float* w2    = (const float*)d_in[4];
  const float* b2    = (const float*)d_in[5];
  const float* w3    = (const float*)d_in[6];
  const float* b3    = (const float*)d_in[7];
  const float* gw1   = (const float*)d_in[8];
  const float* gb1   = (const float*)d_in[9];
  const float* gw2   = (const float*)d_in[10];
  const float* gb2   = (const float*)d_in[11];
  const float* rw1   = (const float*)d_in[12];
  const float* rb1   = (const float*)d_in[13];
  const float* rw2   = (const float*)d_in[14];
  const float* rb2   = (const float*)d_in[15];
  const float* rw3   = (const float*)d_in[16];
  const float* rb3   = (const float*)d_in[17];
  float* out = (float*)d_out;

  const int steps1 = (CIN + 31) / 32;   // 7
  const int steps2 = H1 / 32;           // 16
  const int CP1 = steps1 * 32 + 8;      // 232
  const int CP2 = steps2 * 32 + 8;      // 520

  // workspace layout
  char* ws = (char*)d_ws;
  size_t off = 0;
  auto alloc = [&](size_t bytes) -> char* {
    char* p = ws + off;
    off += (bytes + 255) & ~(size_t)255;
    return p;
  };
  __bf16* pk1   = (__bf16*)alloc((size_t)3 * (H1 / 16) * steps1 * 512 * 2);
  __bf16* pk2   = (__bf16*)alloc((size_t)3 * (H2 / 16) * steps2 * 512 * 2);
  __bf16* pk3   = (__bf16*)alloc((size_t)3 * (FF / 16) * steps2 * 512 * 2);
  __bf16* pgw1  = (__bf16*)alloc((size_t)(GH / 16) * (FF / 32) * 512 * 2);
  float*  stats = (float*) alloc((size_t)2 * BB * 2 * 4);   // stats1 | stats2
  float* stats1 = stats;
  float* stats2 = stats + 2 * BB;
  float*  y1    = (float*) alloc((size_t)BB * H1 * MM * 4);
  float*  y2    = (float*) alloc((size_t)BB * H2 * MM * 4);
  float*  y3    = (float*) alloc((size_t)BB * FF * MM * 4);
  float*  h1    = (float*) alloc((size_t)BB * 128 * 4);
  float*  h2    = (float*) alloc((size_t)BB * 64 * 4);

  hipMemsetAsync(stats, 0, (size_t)2 * BB * 2 * 4, stream);

  // pack weights into WMMA fragment layouts
  {
    int t1 = 3 * (H1 / 16) * steps1 * 512;
    pack_wA<<<(t1 + 255) / 256, 256, 0, stream>>>(w1, pk1, H1, CIN, steps1);
    int t2 = 3 * (H2 / 16) * steps2 * 512;
    pack_wA<<<(t2 + 255) / 256, 256, 0, stream>>>(w2, pk2, H2, H1, steps2);
    int t3 = 3 * (FF / 16) * steps2 * 512;
    pack_wA<<<(t3 + 255) / 256, 256, 0, stream>>>(w3, pk3, FF, H2, steps2);
    int t4 = (GH / 16) * (FF / 32) * 512;
    pack_wB<<<(t4 + 255) / 256, 256, 0, stream>>>(gw1, pgw1, GH, FF);
  }

  const size_t pbytes = (size_t)3 * 64 * 132 * 4;     // 101376
  // layer 1: raw trees, stats -> stats1
  {
    size_t shm = (size_t)128 * CP1 * 2; if (shm < pbytes) shm = pbytes;
    tree_conv_kernel<7><<<BB * (H1 / 64), 256, shm, stream>>>(
        trees, nullptr, idx, pk1, b1, y1, stats1, CIN, H1, 0);
  }
  // layer 2: LN(stats1)+relu on input, stats -> stats2
  {
    size_t shm = (size_t)128 * CP2 * 2; if (shm < pbytes) shm = pbytes;
    tree_conv_kernel<16><<<BB * (H2 / 64), 256, shm, stream>>>(
        y1, stats1, idx, pk2, b2, y2, stats2, H1, H2, H1);
  }
  // layer 3: LN(stats2)+relu on input, no output stats (emb)
  {
    size_t shm = (size_t)128 * CP2 * 2; if (shm < pbytes) shm = pbytes;
    tree_conv_kernel<16><<<BB * (FF / 64), 256, shm, stream>>>(
        y2, stats2, idx, pk3, b3, y3, nullptr, H2, FF, H2);
  }
  // attention pool -> combined written straight into d_out[512:]
  {
    size_t shm = (size_t)128 * 264 * 2 + (size_t)128 * 132 * 4;
    attn_pool_kernel<<<BB, 256, shm, stream>>>(y3, pgw1, gb1, gw2, gb2,
                                               out + BB);
  }
  // readout MLP
  fc_kernel<<<(BB * 128 + 255) / 256, 256, 0, stream>>>(out + BB, rw1, rb1, h1,
                                                        2 * FF, 128, BB, 1);
  fc_kernel<<<(BB * 64 + 255) / 256, 256, 0, stream>>>(h1, rw2, rb2, h2,
                                                       128, 64, BB, 1);
  fc_kernel<<<(BB * 1 + 255) / 256, 256, 0, stream>>>(h2, rw3, rb3, out,
                                                      64, 1, BB, 0);
}
